// dnn_45432164057630
// MI455X (gfx1250) — compile-verified, matrix operand order checked by reading, not run on hardware
//
#include <hip/hip_runtime.h>

#define IN_FEATURES 20000
#define OUT1 192
#define PAIRS 18336          // OUT1*(OUT1-1)/2, = 573*32
#define OUT2 9168            // PAIRS/2, = 573*16
#define OUT2_PAD 9184        // next multiple of 32
#define BATCH 256
#define LN_EPS 1e-5f

typedef __bf16 bf16_t;
typedef __attribute__((ext_vector_type(16))) __bf16 v16bf;
typedef __attribute__((ext_vector_type(8)))  float  v8f;
typedef __attribute__((ext_vector_type(4)))  float  v4f;

// Tell InferAddressSpaces these are global pointers: flat_* -> global_* loads,
// which use only LOADcnt.
__device__ inline void assume_global(const void* p)
{
    __builtin_assume(!__builtin_amdgcn_is_shared((void*)p) &&
                     !__builtin_amdgcn_is_private((void*)p));
}

__device__ inline v16bf zero16()
{
    v16bf r;
    #pragma unroll
    for (int e = 0; e < 16; ++e) r[e] = (bf16_t)0.0f;
    return r;
}

__device__ inline v16bf cvt16(v4f a0, v4f a1, v4f a2, v4f a3)
{
    v16bf r;
    #pragma unroll
    for (int e = 0; e < 4; ++e) {
        r[e]      = (bf16_t)a0[e];
        r[4 + e]  = (bf16_t)a1[e];
        r[8 + e]  = (bf16_t)a2[e];
        r[12 + e] = (bf16_t)a3[e];
    }
    return r;
}

// Load 16 consecutive fp32 (4x b128, distinct temps), convert to bf16.
__device__ inline v16bf load16f_bf16(const float* p)
{
    const v4f* q = (const v4f*)p;
    v4f a0 = q[0], a1 = q[1], a2 = q[2], a3 = q[3];
    return cvt16(a0, a1, a2, a3);
}

// ---------------------------------------------------------------------------
// Generic WMMA GEMM:  C[m,n] (+)= sum_k A[m,k] * W[n,k]
//   A: row-major [M, lda]  (fp32 or bf16; bf16 buffers are zero-padded to lda)
//   W: row-major [N, ldw]  fp32, converted to bf16 in registers
//   Each wave owns a 16 x (16*NT) C tile; optional split-K with atomicAdd.
//   All NT B fragments are materialized before the NT WMMAs so they get
//   distinct register blocks (avoids WMMA->VALU WAR nops on a shared block).
// ---------------------------------------------------------------------------
template<bool A_BF16, bool ATOMIC, int NT>
__global__ __launch_bounds__(256)
void wmma_gemm_kernel(const void* __restrict__ Av, const float* __restrict__ W,
                      float* __restrict__ C,
                      int K, int lda, int ldw, int ldc,
                      int mtiles, int nsupers, int kchunk)
{
    assume_global(Av); assume_global(W); assume_global(C);

    const int lane = threadIdx.x & 31;
    const int gwid = __builtin_amdgcn_readfirstlane(blockIdx.x * 8 + (threadIdx.x >> 5));
    const int ksplits = (K + kchunk - 1) / kchunk;
    if (gwid >= mtiles * nsupers * ksplits) return;   // wave-uniform

    int tmp = gwid;
    const int mt = tmp % mtiles;  tmp /= mtiles;      // mt fastest: W tile L2 reuse
    const int ns = tmp % nsupers; tmp /= nsupers;
    const int ks = tmp;

    const int m     = lane & 15;     // A row / B column within tile
    const int half  = lane >> 4;     // which 16-wide K half this lane holds
    const int nbase = ns * (16 * NT);
    const int k0    = ks * kchunk;
    const int kremq = K - k0;
    const int krem  = (kremq < kchunk) ? kremq : kchunk;
    const int nfull = krem >> 5;                       // guard-free 32-wide steps
    const bool tail16 = (krem & 31) != 0;              // remainder is always 16

    // per-lane (VGPR) element offsets, loop-invariant, 32-bit
    const int a_voff = m * lda + half * 16;
    int w_voff[NT];
    #pragma unroll
    for (int t = 0; t < NT; ++t)
        w_voff[t] = (t * 16 + m) * ldw + half * 16;

    // scalar (SGPR) base pointers, one bump of 32 elements per K-step
    const float*  af = (const float*)Av  + (size_t)(mt * 16) * lda + k0;
    const bf16_t* ab = (const bf16_t*)Av + (size_t)(mt * 16) * lda + k0;
    const float*  wbase = W + (size_t)nbase * ldw + k0;

    v8f acc[NT];
    #pragma unroll
    for (int t = 0; t < NT; ++t) {
        #pragma unroll
        for (int e = 0; e < 8; ++e) acc[t][e] = 0.0f;
    }

    for (int it = 0; it < nfull; ++it) {
        // ---- A fragment: a[i] = A[mrow, k + half*16 + i] ----
        v16bf av;
        if (A_BF16) av = *(const v16bf*)(ab + a_voff);
        else        av = load16f_bf16(af + a_voff);

        __builtin_prefetch(wbase + w_voff[0] + 256, 0, 1);

        // ---- all NT B fragments first (live simultaneously) ----
        v16bf bv[NT];
        #pragma unroll
        for (int t = 0; t < NT; ++t)
            bv[t] = load16f_bf16(wbase + w_voff[t]);

        // ---- then NT WMMAs back-to-back ----
        #pragma unroll
        for (int t = 0; t < NT; ++t)
            acc[t] = __builtin_amdgcn_wmma_f32_16x16x32_bf16(
                false, av, false, bv[t], (short)0, acc[t], false, false);

        af += 32; ab += 32; wbase += 32;
    }

    if (tail16) {   // only GEMM3 (K % 32 == 16): half==1 lanes contribute zeros
        v16bf av;
        if (A_BF16) {
            av = *(const v16bf*)(ab + a_voff);   // bf16 A buffers are zero-padded
        } else {
            if (half == 0) av = load16f_bf16(af + a_voff);
            else           av = zero16();
        }
        v16bf bv[NT];
        #pragma unroll
        for (int t = 0; t < NT; ++t) {
            if (half == 0) bv[t] = load16f_bf16(wbase + w_voff[t]);
            else           bv[t] = zero16();
        }
        #pragma unroll
        for (int t = 0; t < NT; ++t)
            acc[t] = __builtin_amdgcn_wmma_f32_16x16x32_bf16(
                false, av, false, bv[t], (short)0, acc[t], false, false);
    }

    // ---- C writeback: c[v] -> C[mt*16 + half*8 + v, ncol] ----
    #pragma unroll
    for (int t = 0; t < NT; ++t) {
        const int ncol = nbase + t * 16 + m;
        #pragma unroll
        for (int v = 0; v < 8; ++v) {
            const int row = mt * 16 + half * 8 + v;
            if (ATOMIC) atomicAdd(&C[(size_t)row * ldc + ncol], acc[t][v]);
            else        C[(size_t)row * ldc + ncol] = acc[t][v];
        }
    }
}

// ---------------------------------------------------------------------------
// Block reduction helper (256 threads): returns (sum_a, sum_b) to all threads
// ---------------------------------------------------------------------------
__device__ inline float2 block_reduce2(float a, float b, float* red, float* red2)
{
    const int tid = threadIdx.x;
    red[tid] = a; red2[tid] = b;
    __syncthreads();
    for (int off = 128; off > 0; off >>= 1) {
        if (tid < off) { red[tid] += red[tid + off]; red2[tid] += red2[tid + off]; }
        __syncthreads();
    }
    float2 r; r.x = red[0]; r.y = red2[0];
    __syncthreads();   // safe reuse of red arrays
    return r;
}

// ---------------------------------------------------------------------------
// Pair-ratio expansion + LayerNorm over PAIRS, output bf16
// One block (256 threads) per batch row.
// ---------------------------------------------------------------------------
__global__ __launch_bounds__(256)
void pair_ln_kernel(const float* __restrict__ h,
                    const float* __restrict__ g1, const float* __restrict__ b1,
                    bf16_t* __restrict__ r)
{
    assume_global(h); assume_global(g1); assume_global(b1); assume_global(r);

    __shared__ float y[OUT1];
    __shared__ int   S[OUT1];     // S[i] = #pairs with first index < i
    __shared__ float red[256], red2[256];

    const int b = blockIdx.x, tid = threadIdx.x;
    for (int i = tid; i < OUT1; i += 256) {
        y[i] = h[(size_t)b * OUT1 + i] + 1.0f;
        S[i] = i * (OUT1 - 1) - (i * (i - 1)) / 2;
    }
    __syncthreads();

    float s = 0.0f, ss = 0.0f;
    for (int p = tid; p < PAIRS; p += 256) {
        int lo = 0, hi = OUT1 - 2;                     // i in [0, 190]
        while (lo < hi) { int mid = (lo + hi + 1) >> 1;
                          if (S[mid] <= p) lo = mid; else hi = mid - 1; }
        const int i = lo, j = i + 1 + (p - S[i]);
        const float rr = y[j] / y[i];
        s += rr; ss += rr * rr;
    }
    float2 red_out = block_reduce2(s, ss, red, red2);
    const float mean = red_out.x * (1.0f / PAIRS);
    const float var  = red_out.y * (1.0f / PAIRS) - mean * mean;
    const float rs   = rsqrtf(var + LN_EPS);

    for (int p = tid; p < PAIRS; p += 256) {
        int lo = 0, hi = OUT1 - 2;
        while (lo < hi) { int mid = (lo + hi + 1) >> 1;
                          if (S[mid] <= p) lo = mid; else hi = mid - 1; }
        const int i = lo, j = i + 1 + (p - S[i]);
        const float rr = y[j] / y[i];
        const float v  = (rr - mean) * rs * g1[p] + b1[p];
        r[(size_t)b * PAIRS + p] = (bf16_t)v;
    }
}

// ---------------------------------------------------------------------------
// ReLU + LayerNorm, output bf16 zero-padded to ldo columns
// ---------------------------------------------------------------------------
__global__ __launch_bounds__(256)
void relu_ln_bf16_kernel(const float* __restrict__ in,
                         const float* __restrict__ g, const float* __restrict__ bp,
                         bf16_t* __restrict__ out, int N, int ldo)
{
    assume_global(in); assume_global(g); assume_global(bp); assume_global(out);

    __shared__ float red[256], red2[256];
    const int b = blockIdx.x, tid = threadIdx.x;

    float s = 0.0f, ss = 0.0f;
    for (int o = tid; o < N; o += 256) {
        float v = in[(size_t)b * N + o];
        v = v > 0.0f ? v : 0.0f;
        s += v; ss += v * v;
    }
    float2 red_out = block_reduce2(s, ss, red, red2);
    const float mean = red_out.x / (float)N;
    const float var  = red_out.y / (float)N - mean * mean;
    const float rs   = rsqrtf(var + LN_EPS);

    for (int o = tid; o < ldo; o += 256) {
        if (o < N) {
            float v = in[(size_t)b * N + o];
            v = v > 0.0f ? v : 0.0f;
            out[(size_t)b * ldo + o] = (bf16_t)((v - mean) * rs * g[o] + bp[o]);
        } else {
            out[(size_t)b * ldo + o] = (bf16_t)0.0f;   // pad for K%32 tail
        }
    }
}

// ---------------------------------------------------------------------------
// ReLU + LayerNorm + dot with Wout + bout -> out[b]
// ---------------------------------------------------------------------------
__global__ __launch_bounds__(256)
void relu_ln_dot_kernel(const float* __restrict__ in,
                        const float* __restrict__ g, const float* __restrict__ bp,
                        const float* __restrict__ Wout, const float* __restrict__ bout,
                        float* __restrict__ out, int N)
{
    assume_global(in); assume_global(g); assume_global(bp);
    assume_global(Wout); assume_global(bout); assume_global(out);

    __shared__ float red[256], red2[256];
    const int b = blockIdx.x, tid = threadIdx.x;

    float s = 0.0f, ss = 0.0f;
    for (int o = tid; o < N; o += 256) {
        float v = in[(size_t)b * N + o];
        v = v > 0.0f ? v : 0.0f;
        s += v; ss += v * v;
    }
    float2 red_out = block_reduce2(s, ss, red, red2);
    const float mean = red_out.x / (float)N;
    const float var  = red_out.y / (float)N - mean * mean;
    const float rs   = rsqrtf(var + LN_EPS);

    float dot = 0.0f;
    for (int o = tid; o < N; o += 256) {
        float v = in[(size_t)b * N + o];
        v = v > 0.0f ? v : 0.0f;
        dot += ((v - mean) * rs * g[o] + bp[o]) * Wout[o];
    }
    float2 d = block_reduce2(dot, 0.0f, red, red2);
    if (tid == 0) out[b] = d.x + bout[0];
}

// ---------------------------------------------------------------------------
extern "C" void kernel_launch(void* const* d_in, const int* in_sizes, int n_in,
                              void* d_out, int out_size, void* d_ws, size_t ws_size,
                              hipStream_t stream)
{
    (void)in_sizes; (void)n_in; (void)out_size; (void)ws_size;
    const float* x    = (const float*)d_in[0];
    const float* W1   = (const float*)d_in[1];
    const float* g1   = (const float*)d_in[2];
    const float* b1   = (const float*)d_in[3];
    const float* W2   = (const float*)d_in[4];
    const float* g2   = (const float*)d_in[5];
    const float* b2   = (const float*)d_in[6];
    const float* W3   = (const float*)d_in[7];
    const float* g3   = (const float*)d_in[8];
    const float* b3   = (const float*)d_in[9];
    const float* Wout = (const float*)d_in[10];
    const float* bout = (const float*)d_in[11];
    float* out = (float*)d_out;

    char* ws = (char*)d_ws;
    size_t off = 0;
    auto take = [&](size_t bytes) {
        size_t p = off; off += (bytes + 255) & ~(size_t)255; return p;
    };
    float*  h     = (float*) (ws + take((size_t)BATCH * OUT1 * 4));
    bf16_t* rbuf  = (bf16_t*)(ws + take((size_t)BATCH * PAIRS * 2));
    float*  h2pre = (float*) (ws + take((size_t)BATCH * OUT2 * 4));
    bf16_t* h2bf  = (bf16_t*)(ws + take((size_t)BATCH * OUT2_PAD * 2));
    float*  h3pre = (float*) (ws + take((size_t)BATCH * OUT2 * 4));

    // GEMM1 uses split-K atomics -> zero the accumulator each call
    (void)hipMemsetAsync(h, 0, (size_t)BATCH * OUT1 * 4, stream);

    // ---- GEMM1: h = x @ W1^T  [256,192], split-K (25 chunks of 800) ----
    {
        const int mt = BATCH / 16, ns = OUT1 / 48, ksplits = 25, kchunk = 800;
        const int blocks = (mt * ns * ksplits + 7) / 8;
        wmma_gemm_kernel<false, true, 3><<<blocks, 256, 0, stream>>>(
            x, W1, h, IN_FEATURES,
            IN_FEATURES, IN_FEATURES, OUT1, mt, ns, kchunk);
    }

    // ---- pair expansion + LN1 -> bf16 r [256, 18336] ----
    pair_ln_kernel<<<BATCH, 256, 0, stream>>>(h, g1, b1, rbuf);

    // ---- GEMM2: h2pre = r @ W2^T  [256, 9168], K = 18336 ----
    {
        const int mt = BATCH / 16, ns = OUT2 / 48;
        const int blocks = (mt * ns + 7) / 8;
        wmma_gemm_kernel<true, false, 3><<<blocks, 256, 0, stream>>>(
            rbuf, W2, h2pre, PAIRS,
            PAIRS, PAIRS, OUT2, mt, ns, PAIRS);
    }

    // ---- ReLU + LN2 -> bf16 h2 [256, 9184 padded] ----
    relu_ln_bf16_kernel<<<BATCH, 256, 0, stream>>>(h2pre, g2, b2, h2bf, OUT2, OUT2_PAD);

    // ---- GEMM3: h3pre = h2 @ W3^T  [256, 9168], K = 9168 (tail step) ----
    {
        const int mt = BATCH / 16, ns = OUT2 / 48;
        const int blocks = (mt * ns + 7) / 8;
        wmma_gemm_kernel<true, false, 3><<<blocks, 256, 0, stream>>>(
            h2bf, W3, h3pre, OUT2,
            OUT2_PAD, OUT2, OUT2, mt, ns, OUT2_PAD);
    }

    // ---- ReLU + LN3 + output head ----
    relu_ln_dot_kernel<<<BATCH, 256, 0, stream>>>(h3pre, g3, b3, Wout, bout, out, OUT2);
}
